// SConvNet_36696200577233
// MI455X (gfx1250) — compile-verified, matrix operand order checked by reading, not run on hardware
//
#include <hip/hip_runtime.h>
#include <hip/hip_bf16.h>

// ---------------------------------------------------------------------------
// SConvNet forward for MI455X (gfx1250, wave32, WMMA).
// GEMMs: v_wmma_f32_16x16x32_bf16, 128x128 block tile, 8 waves,
//        async global->LDS tile staging (ASYNCcnt path).
// sconv's FFT is replaced by its equivalent first-order complex recurrence.
// ---------------------------------------------------------------------------

#define DEPTH 4
#define DIM   512
#define DFF   2048
#define VOCAB 4096
#define BATCH 2
#define SEQ   4096
#define MTOT  (BATCH * SEQ)   // 8192 rows of activations

typedef __attribute__((ext_vector_type(16))) __bf16 v16bf;
typedef __attribute__((ext_vector_type(8)))  __bf16 v8bf;
typedef __attribute__((ext_vector_type(8)))  float  v8f;

// ---------------------------------------------------------------------------
// f32 -> bf16 conversion (grid-stride)
// ---------------------------------------------------------------------------
__global__ __launch_bounds__(256) void k_cvt_bf16(const float* __restrict__ in,
                                                  __bf16* __restrict__ out,
                                                  size_t n) {
  size_t i = (size_t)blockIdx.x * blockDim.x + threadIdx.x;
  size_t stride = (size_t)gridDim.x * blockDim.x;
  for (; i < n; i += stride) out[i] = (__bf16)in[i];
}

// ---------------------------------------------------------------------------
// Async 16B global -> LDS copy (GLOBAL_LOAD_ASYNC_TO_LDS_B128, ASYNCcnt).
// Per ISA §15.18.3: each lane copies 16B from its global address to its LDS offset.
// LDS generic pointers carry the LDS byte offset in the low 32 bits.
// ---------------------------------------------------------------------------
__device__ inline void async_copy16(unsigned int lds_off, const void* gptr) {
  asm volatile("global_load_async_to_lds_b128 %0, %1, off"
               :: "v"(lds_off), "v"((unsigned long long)(uintptr_t)gptr)
               : "memory");
}
__device__ inline void wait_async0() {
  asm volatile("s_wait_asynccnt 0x0" ::: "memory");
}

// ---------------------------------------------------------------------------
// Build a 16-element bf16 WMMA fragment from two 16-byte chunks.
// Per ISA §7.12.2 (16-bit A 16x32): lane L holds row L%16; VGPR0..3 hold
// K = (L<16 ? 0..7 : 8..15), VGPR4..7 hold K = (L<16 ? 16..23 : 24..31).
// ---------------------------------------------------------------------------
__device__ inline v16bf frag_cat(v8bf lo, v8bf hi) {
  v16bf r;
#pragma unroll
  for (int i = 0; i < 8; ++i) { r[i] = lo[i]; r[i + 8] = hi[i]; }
  return r;
}

__device__ inline float silu_f(float v) { return v / (1.0f + expf(-v)); }

// ---------------------------------------------------------------------------
// TN GEMM: out[M,N] = epilogue( A[M,K] * W[N,K]^T + bias[N] )
//   - 256 threads = 8 waves; block tile 128(M) x 128(N); K-step 32
//   - wave (wm = wid&3, wn = wid>>2) computes 32(M) x 64(N):
//       2 A-fragments x 4 B-fragments = 8 WMMA per K-step, 12 ds_load_b128
//   - A,B tiles staged in LDS via async global->LDS copies
//     (row stride padded to 40 bf16 -> conflict-free ds_load_b128 reads)
// epilogue flags: SILU, RES (out += existing outF), OUTF (write f32),
//                 OUTB (write bf16)
// ---------------------------------------------------------------------------
#define LDS_STRIDE 40

template <bool SILU, bool RES, bool OUTF, bool OUTB>
__global__ __launch_bounds__(256) void k_gemm(const __bf16* __restrict__ A,
                                              const __bf16* __restrict__ W,
                                              const float* __restrict__ bias,
                                              float* __restrict__ outF,
                                              __bf16* __restrict__ outB,
                                              int N, int K) {
  __shared__ __bf16 sA[128 * LDS_STRIDE];
  __shared__ __bf16 sB[128 * LDS_STRIDE];

  const int tid  = threadIdx.x;
  const int lane = tid & 31;
  const int wid  = tid >> 5;
  const int wm   = wid & 3;    // M quarter (32 rows)
  const int wn   = wid >> 2;   // N half    (64 cols)
  const int bm   = blockIdx.x * 128;
  const int bn   = blockIdx.y * 128;

  const unsigned int sAo = (unsigned int)(uintptr_t)&sA[0];
  const unsigned int sBo = (unsigned int)(uintptr_t)&sB[0];

  v8f acc[2][4] = {};

  const int rA0 = (wm << 5) + (lane & 15);   // first A-frag LDS row
  const int kc  = (lane < 16) ? 0 : 8;       // K sub-chunk per half-wave

  for (int k0 = 0; k0 < K; k0 += 32) {
    // --- async stage of 128x32 bf16 tiles of A and W into LDS ---
#pragma unroll
    for (int it = 0; it < 2; ++it) {
      const int c   = tid + it * 256;   // 512 16B-chunks per tile
      const int row = c >> 2;
      const int sub = c & 3;
      const unsigned int loff = (unsigned int)(row * LDS_STRIDE + sub * 8) * 2u;
      async_copy16(sAo + loff, A + (long)(bm + row) * K + k0 + sub * 8);
      async_copy16(sBo + loff, W + (long)(bn + row) * K + k0 + sub * 8);
    }
    if (k0 + 32 < K) {  // global_prefetch_b8 of next K tile
      __builtin_prefetch(A + (long)(bm + (tid >> 1)) * K + k0 + 32, 0, 1);
      __builtin_prefetch(W + (long)(bn + (tid >> 1)) * K + k0 + 32, 0, 1);
    }
    wait_async0();
    __syncthreads();

    // --- fragments + 8 WMMAs per wave ---
    v16bf a0 = frag_cat(*(const v8bf*)(sA + rA0 * LDS_STRIDE + kc),
                        *(const v8bf*)(sA + rA0 * LDS_STRIDE + kc + 16));
    v16bf a1 = frag_cat(*(const v8bf*)(sA + (rA0 + 16) * LDS_STRIDE + kc),
                        *(const v8bf*)(sA + (rA0 + 16) * LDS_STRIDE + kc + 16));
#pragma unroll
    for (int j = 0; j < 4; ++j) {
      const int rB = (wn << 6) + (j << 4) + (lane & 15);
      v16bf b = frag_cat(*(const v8bf*)(sB + rB * LDS_STRIDE + kc),
                         *(const v8bf*)(sB + rB * LDS_STRIDE + kc + 16));
      acc[0][j] = __builtin_amdgcn_wmma_f32_16x16x32_bf16(
          false, a0, false, b, (short)0, acc[0][j], false, false);
      acc[1][j] = __builtin_amdgcn_wmma_f32_16x16x32_bf16(
          false, a1, false, b, (short)0, acc[1][j], false, false);
    }
    __syncthreads();
  }

  // --- epilogue. C layout (§7.12.2): lane L, VGPR r -> M = r + (L>=16?8:0),
  // N = L%16 ---
  const int nLoc = lane & 15;
#pragma unroll
  for (int mi = 0; mi < 2; ++mi) {
    const int mBase = bm + (wm << 5) + (mi << 4) + ((lane & 16) >> 1);
#pragma unroll
    for (int j = 0; j < 4; ++j) {
      const int col = bn + (wn << 6) + (j << 4) + nLoc;
      const float bv = bias[col];
#pragma unroll
      for (int r = 0; r < 8; ++r) {
        const long idx = (long)(mBase + r) * N + col;
        float v = acc[mi][j][r] + bv;
        if (SILU) v = silu_f(v);
        if (RES)  v += outF[idx];
        if (OUTF) outF[idx] = v;
        if (OUTB) outB[idx] = (__bf16)v;
      }
    }
  }
}

// ---------------------------------------------------------------------------
// LayerNorm over DIM=512, one row per block (128 threads, 4 elems/thread),
// wave32 shuffle reduction + cross-wave LDS. Optional SiLU / f32 / bf16 out.
// ---------------------------------------------------------------------------
template <bool SILU, bool OUTF, bool OUTB>
__global__ __launch_bounds__(128) void k_ln(const float* __restrict__ in,
                                            const float* __restrict__ w,
                                            const float* __restrict__ b,
                                            float* __restrict__ outF,
                                            __bf16* __restrict__ outB) {
  const int row = blockIdx.x;
  const int tid = threadIdx.x;
  const float* p = in + (long)row * DIM;

  float v[4];
  float s = 0.f, s2 = 0.f;
#pragma unroll
  for (int i = 0; i < 4; ++i) {
    v[i] = p[tid + i * 128];
    s += v[i];
    s2 += v[i] * v[i];
  }
#pragma unroll
  for (int o = 16; o; o >>= 1) {
    s  += __shfl_xor(s, o, 32);
    s2 += __shfl_xor(s2, o, 32);
  }
  __shared__ float rs[4], rq[4];
  const int wid = tid >> 5, lane = tid & 31;
  if (lane == 0) { rs[wid] = s; rq[wid] = s2; }
  __syncthreads();
  s  = rs[0] + rs[1] + rs[2] + rs[3];
  s2 = rq[0] + rq[1] + rq[2] + rq[3];
  const float mu   = s * (1.0f / DIM);
  const float var  = s2 * (1.0f / DIM) - mu * mu;
  const float rstd = rsqrtf(var + 1e-5f);

#pragma unroll
  for (int i = 0; i < 4; ++i) {
    const int c = tid + i * 128;
    float o = (v[i] - mu) * rstd * w[c] + b[c];
    if (SILU) o = silu_f(o);
    if (OUTF) outF[(long)row * DIM + c] = o;
    if (OUTB) outB[(long)row * DIM + c] = (__bf16)o;
  }
}

// ---------------------------------------------------------------------------
// sconv: complex first-order recurrence c[t] = p*c[t-1] + y[t], c[-1]=lc,
// p = tanh(|phz|) * phz/|phz|.  Fused residual: h[b,t,d] += Re(c[t])*scl[d].
// One thread per (b,d) channel; lanes cover consecutive d -> coalesced.
// ---------------------------------------------------------------------------
__global__ __launch_bounds__(128) void k_sconv(const float* __restrict__ y,
                                               float* __restrict__ h,
                                               const float* __restrict__ pr,
                                               const float* __restrict__ pi,
                                               const float* __restrict__ lr,
                                               const float* __restrict__ li,
                                               const float* __restrict__ scl) {
  const int tid = blockIdx.x * blockDim.x + threadIdx.x;
  if (tid >= BATCH * DIM) return;
  const int bb = tid / DIM;
  const int d  = tid % DIM;

  const float ar = pr[d], ai = pi[d];
  const float r  = sqrtf(ar * ar + ai * ai);
  const float sc = tanhf(r) / fmaxf(r, 1e-30f);
  const float p_re = sc * ar, p_im = sc * ai;

  float c_re = lr[d], c_im = li[d];
  const float g = scl[d];

  const float* yp = y + (long)bb * SEQ * DIM + d;
  float*       hp = h + (long)bb * SEQ * DIM + d;
#pragma unroll 4
  for (int t = 0; t < SEQ; ++t) {
    const float xv = yp[(long)t * DIM];
    const float nr = p_re * c_re - p_im * c_im + xv;
    const float ni = p_re * c_im + p_im * c_re;
    c_re = nr;
    c_im = ni;
    hp[(long)t * DIM] += c_re * g;
  }
}

// ---------------------------------------------------------------------------
// Host-side orchestration
// ---------------------------------------------------------------------------
static inline size_t align256(size_t x) { return (x + 255) & ~(size_t)255; }

extern "C" void kernel_launch(void* const* d_in, const int* in_sizes, int n_in,
                              void* d_out, int out_size, void* d_ws, size_t ws_size,
                              hipStream_t stream) {
  (void)in_sizes; (void)n_in; (void)out_size; (void)ws_size;

  const float* x     = (const float*)d_in[0];
  const float* Win   = (const float*)d_in[1];
  const float* bin   = (const float*)d_in[2];
  const float* ln1_w = (const float*)d_in[3];
  const float* ln1_b = (const float*)d_in[4];
  const float* phz_r = (const float*)d_in[5];
  const float* phz_i = (const float*)d_in[6];
  const float* lc_r  = (const float*)d_in[7];
  const float* lc_i  = (const float*)d_in[8];
  const float* sclin = (const float*)d_in[9];
  const float* ln2_w = (const float*)d_in[10];
  const float* ln2_b = (const float*)d_in[11];
  const float* W1    = (const float*)d_in[12];
  const float* b1    = (const float*)d_in[13];
  const float* W2    = (const float*)d_in[14];
  const float* b2    = (const float*)d_in[15];
  const float* lnl_w = (const float*)d_in[16];
  const float* lnl_b = (const float*)d_in[17];
  const float* Wout  = (const float*)d_in[18];
  const float* bout  = (const float*)d_in[19];
  float* out = (float*)d_out;

  // ---- workspace layout ----
  char* ws = (char*)d_ws;
  size_t off = 0;
  const size_t nX    = (size_t)MTOT * VOCAB;
  const size_t nWin  = (size_t)DIM * VOCAB;
  const size_t nW1   = (size_t)DEPTH * DFF * DIM;
  const size_t nW2   = (size_t)DEPTH * DIM * DFF;
  const size_t nWout = (size_t)VOCAB * DIM;

  __bf16* x_bf    = (__bf16*)(ws + off); off = align256(off + nX * 2);
  __bf16* win_bf  = (__bf16*)(ws + off); off = align256(off + nWin * 2);
  __bf16* w1_bf   = (__bf16*)(ws + off); off = align256(off + nW1 * 2);
  __bf16* w2_bf   = (__bf16*)(ws + off); off = align256(off + nW2 * 2);
  __bf16* wout_bf = (__bf16*)(ws + off); off = align256(off + nWout * 2);
  float*  h       = (float*)(ws + off);  off = align256(off + (size_t)MTOT * DIM * 4);
  float*  y       = (float*)(ws + off);  off = align256(off + (size_t)MTOT * DIM * 4);
  __bf16* ln_bf   = (__bf16*)(ws + off); off = align256(off + (size_t)MTOT * DIM * 2);
  __bf16* mid_bf  = (__bf16*)(ws + off); off = align256(off + (size_t)MTOT * DFF * 2);

  // ---- convert GEMM operands to bf16 ----
  auto cvt = [&](const float* src, __bf16* dst, size_t n) {
    int blocks = (int)((n + 255) / 256);
    if (blocks > 4096) blocks = 4096;
    k_cvt_bf16<<<blocks, 256, 0, stream>>>(src, dst, n);
  };
  cvt(x, x_bf, nX);
  cvt(Win, win_bf, nWin);
  cvt(W1, w1_bf, nW1);
  cvt(W2, w2_bf, nW2);
  cvt(Wout, wout_bf, nWout);

  // ---- token_in: h = x @ Win^T + bin  (M=8192, N=512, K=4096) ----
  {
    dim3 g(MTOT / 128, DIM / 128);
    k_gemm<false, false, true, false><<<g, 256, 0, stream>>>(
        x_bf, win_bf, bin, h, nullptr, DIM, VOCAB);
  }

  // ---- layers ----
  for (int l = 0; l < DEPTH; ++l) {
    // y = silu(LN1(h))
    k_ln<true, true, false><<<MTOT, 128, 0, stream>>>(
        h, ln1_w + l * DIM, ln1_b + l * DIM, y, nullptr);
    // h += sconv(y) * sc_lin
    k_sconv<<<(BATCH * DIM + 127) / 128, 128, 0, stream>>>(
        y, h, phz_r + l * DIM, phz_i + l * DIM, lc_r + l * DIM,
        lc_i + l * DIM, sclin + l * DIM);
    // ln_bf = LN2(h) (bf16)
    k_ln<false, false, true><<<MTOT, 128, 0, stream>>>(
        h, ln2_w + l * DIM, ln2_b + l * DIM, nullptr, ln_bf);
    // mid = silu(ln_bf @ W1^T + b1)  (M=8192, N=2048, K=512) -> bf16
    {
      dim3 g(MTOT / 128, DFF / 128);
      k_gemm<true, false, false, true><<<g, 256, 0, stream>>>(
          ln_bf, w1_bf + (size_t)l * DFF * DIM, b1 + l * DFF,
          nullptr, mid_bf, DFF, DIM);
    }
    // h += mid @ W2^T + b2  (M=8192, N=512, K=2048)
    {
      dim3 g(MTOT / 128, DIM / 128);
      k_gemm<false, true, true, false><<<g, 256, 0, stream>>>(
          mid_bf, w2_bf + (size_t)l * DIM * DFF, b2 + l * DIM,
          h, nullptr, DIM, DFF);
    }
  }

  // ---- final LN + vocab projection ----
  k_ln<false, false, true><<<MTOT, 128, 0, stream>>>(
      h, lnl_w, lnl_b, nullptr, ln_bf);
  {
    dim3 g(MTOT / 128, VOCAB / 128);
    k_gemm<false, false, true, false><<<g, 256, 0, stream>>>(
        ln_bf, wout_bf, bout, out, nullptr, VOCAB, DIM);
  }
}
